// TinyNoiseGru_28810640622007
// MI455X (gfx1250) — matrix operand order repaired and checked by reading, not compiled, not run on hardware
//
#include <hip/hip_runtime.h>

#define B_TOT 128
#define T_LEN 4096
#define FEATD 64
#define HID   24
#define NF    40   // LDS-resident B fragments

typedef __attribute__((ext_vector_type(16))) _Float16 v16h;
typedef __attribute__((ext_vector_type(8)))  float    v8f;

struct __attribute__((aligned(16))) U4 { unsigned int a, b, c, d; };
struct __attribute__((aligned(8)))  U2 { unsigned int a, b; };

union AFrag { _Float16 h[16]; v16h v; U4 q[2]; };
union CFrag { float f[8]; v8f v; };
union PK    { _Float16 h[4]; U2 u; };

__device__ __forceinline__ v8f wmma_f16(v16h a, v16h b, v8f c) {
  return __builtin_amdgcn_wmma_f32_16x16x32_f16(
      false, a, false, b, (short)0, c, false, false);
}

__device__ __forceinline__ v16h ldfrag(const U4* q) {
  AFrag a; a.q[0] = q[0]; a.q[1] = q[1]; return a.v;
}
__device__ __forceinline__ v16h mkfrag(U4 q0, U4 q1) {
  AFrag a; a.q[0] = q0; a.q[1] = q1; return a.v;
}

__device__ __forceinline__ float sigf(float v) { return __frcp_rn(1.f + __expf(-v)); }
__device__ __forceinline__ float tanhfast(float v) { return 2.f * sigf(2.f * v) - 1.f; }

__global__ __launch_bounds__(32)
void tiny_gru_persist(const float* __restrict__ x,
                      const float* __restrict__ h1in,
                      const float* __restrict__ h2in,
                      const float* __restrict__ convw,
                      const float* __restrict__ convb,
                      const float* __restrict__ gam,
                      const float* __restrict__ bet,
                      const float* __restrict__ wih1,
                      const float* __restrict__ whh1,
                      const float* __restrict__ bih1,
                      const float* __restrict__ bhh1,
                      const float* __restrict__ wih2,
                      const float* __restrict__ whh2,
                      const float* __restrict__ bih2,
                      const float* __restrict__ bhh2,
                      const float* __restrict__ fcw,
                      const float* __restrict__ fcb,
                      float* __restrict__ out)
{
  // f16 B fragments in exact WMMA B layout, 32B per lane each:
  // [0..7] conv (4g x 2ck) [8..19] w_ih1 (6tn x 2ck)
  // [20..25] w_hh1 [26..31] w_ih2 [32..37] w_hh2 [38..39] fc
  __shared__ U4 frag[NF][32][2];
  __shared__ U4 xq[4][16][8];      // packed-f16 x window; slice 3 = zeros (causal pad)
  __shared__ float4 ybq[16][16];   // conv output f32
  __shared__ U4 yhq[16][8];        // normalized y, packed f16
  __shared__ U4 hb1q[16][4];       // h1 state, packed f16, cols 24..31 pad
  __shared__ U4 hb2q[16][4];
  __shared__ float4 nbq[16][8];    // fc output staging
  __shared__ float4 sgamq[16], sbetq[16];

  const int lane = threadIdx.x;
  const int hi = lane >> 4;
  const int nl = lane & 15;
  const int b0 = blockIdx.x << 4;

  const size_t NOISE_N  = (size_t)B_TOT * T_LEN * 32;
  const size_t H1_OFF   = NOISE_N;
  const size_t H2_OFF   = NOISE_N + (size_t)B_TOT * HID;
  const size_t CONF_OFF = NOISE_N + 2 * (size_t)B_TOT * HID;

  // ---------------- setup: gamma/beta as float4 rows ------------------------
  {
    float* g4 = (float*)sgamq; float* b4 = (float*)sbetq;
    for (int i = lane; i < FEATD; i += 32) { g4[i] = gam[i]; b4[i] = bet[i]; }
  }
  // ---------------- setup: per-lane register constants ----------------------
  float rbi1[6], rbh1[6], rbi2[6], rbh2[6];
  #pragma unroll
  for (int tn = 0; tn < 6; ++tn) {
    int npad = tn * 16 + nl, g = npad >> 5, u = npad & 31;
    int idx = g * HID + (u < HID ? u : 0);
    float msk = (u < HID) ? 1.f : 0.f;
    rbi1[tn] = bih1[idx] * msk; rbh1[tn] = bhh1[idx] * msk;
    rbi2[tn] = bih2[idx] * msk; rbh2[tn] = bhh2[idx] * msk;
  }
  float cbv[4];
  #pragma unroll
  for (int g = 0; g < 4; ++g) cbv[g] = convb[g * 16 + nl];
  const float fcb0 = fcb[nl], fcb1 = fcb[16 + nl];

  // ---------------- setup: conv B fragments --------------------------------
  for (int g = 0; g < 4; ++g)
    for (int ck = 0; ck < 2; ++ck) {
      AFrag a;
      #pragma unroll
      for (int j = 0; j < 16; ++j) {
        int k48 = ck * 32 + (j < 8 ? hi * 8 + j : 16 + hi * 8 + (j - 8));
        float v = 0.f;
        if (k48 < 48) { int kk = k48 >> 4, ii = k48 & 15; v = convw[(g * 16 + nl) * 48 + ii * 3 + kk]; }
        a.h[j] = (_Float16)v;
      }
      frag[g * 2 + ck][lane][0] = a.q[0]; frag[g * 2 + ck][lane][1] = a.q[1];
    }
  // ---------------- setup: gated weight B fragments ------------------------
  auto buildGated = [&](const float* W, int Kv, int CK, int base) {
    for (int tn = 0; tn < 6; ++tn)
      for (int ck = 0; ck < CK; ++ck) {
        int npad = tn * 16 + nl, g = npad >> 5, u = npad & 31;
        AFrag a;
        #pragma unroll
        for (int j = 0; j < 16; ++j) {
          int k = ck * 32 + (j < 8 ? hi * 8 + j : 16 + hi * 8 + (j - 8));
          float v = 0.f;
          if (u < HID && k < Kv) v = W[(g * HID + u) * Kv + k];
          a.h[j] = (_Float16)v;
        }
        int idx = base + tn * CK + ck;
        frag[idx][lane][0] = a.q[0]; frag[idx][lane][1] = a.q[1];
      }
  };
  buildGated(wih1, FEATD, 2, 8);
  buildGated(whh1, HID, 1, 20);
  buildGated(wih2, HID, 1, 26);
  buildGated(whh2, HID, 1, 32);
  for (int tn = 0; tn < 2; ++tn) {
    AFrag a;
    #pragma unroll
    for (int j = 0; j < 16; ++j) {
      int k = (j < 8 ? hi * 8 + j : 16 + hi * 8 + (j - 8));
      a.h[j] = (_Float16)((k < HID) ? fcw[(tn * 16 + nl) * HID + k] : 0.f);
    }
    frag[38 + tn][lane][0] = a.q[0]; frag[38 + tn][lane][1] = a.q[1];
  }

  // ---------------- setup: zero pad slice + hidden state -------------------
  {
    const U4 z4 = {0u, 0u, 0u, 0u};
    U4* zp = &xq[3][0][0];
    for (int i = lane; i < 128; i += 32) zp[i] = z4;
  }
  CFrag h1c[2], h2c[2];
  #pragma unroll
  for (int tile = 0; tile < 2; ++tile)
    #pragma unroll
    for (int r = 0; r < 8; ++r) {
      int u = tile * 16 + nl;
      int uc = (u < HID) ? u : 0;
      float msk = (u < HID) ? 1.f : 0.f;
      int b = b0 + r + (hi << 3);
      h1c[tile].f[r] = h1in[b * HID + uc] * msk;
      h2c[tile].f[r] = h2in[b * HID + uc] * msk;
    }
  {
    _Float16* p1 = (_Float16*)hb1q;
    _Float16* p2 = (_Float16*)hb2q;
    int col = lane & 31;
    int cc = (col < HID) ? col : 0;
    float msk = (col < HID) ? 1.f : 0.f;
    for (int r = 0; r < 16; ++r) {
      p1[r * 32 + col] = (_Float16)(h1in[(b0 + r) * HID + cc] * msk);
      p2[r * 32 + col] = (_Float16)(h2in[(b0 + r) * HID + cc] * msk);
    }
  }

  const U4* fb = &frag[0][lane][0];   // fragment i at fb + i*64 (U4 units)
  const U4  ZQ = {0u, 0u, 0u, 0u};
  float* yb = (float*)ybq;
  float* nb = (float*)nbq;
  _Float16* h1p = (_Float16*)hb1q;
  _Float16* h2p = (_Float16*)hb2q;

  int s0 = 3, s1 = 3, s2 = 0;  // rotating time-slice ids (3 = zero slice)

  // ======================= sequential scan over T ==========================
  for (int t = 0; t < T_LEN; ++t) {
    // ---- load x tile [16b x 64f], f32 -> packed f16, coalesced ------------
    #pragma unroll
    for (int i = 0; i < 8; ++i) {
      int fl = i * 128 + lane * 4;
      int b = fl >> 6, f = fl & 63;
      const float4 xv = *(const float4*)(x + ((size_t)(b0 + b) * T_LEN + t) * FEATD + f);
      PK p;
      p.h[0] = (_Float16)xv.x; p.h[1] = (_Float16)xv.y;
      p.h[2] = (_Float16)xv.z; p.h[3] = (_Float16)xv.w;
      ((U2*)&xq[s2][b][0])[f >> 2] = p.u;
    }
    if (t + 8 < T_LEN)
      __builtin_prefetch(x + ((size_t)(b0 + nl) * T_LEN + (t + 8)) * FEATD + hi * 32, 0, 1);

    // ---- grouped causal conv: 4 x (K=64) WMMA GEMMs -----------------------
    {
      const U4* xr0 = &xq[s0][nl][0];
      const U4* xr1 = &xq[s1][nl][0];
      const U4* xr2 = &xq[s2][nl][0];
      #pragma unroll
      for (int g = 0; g < 4; ++g) {
        v16h a0 = mkfrag(xr0[g * 2 + hi], xr1[g * 2 + hi]); // K 0..31 (t-2, t-1)
        v16h a1 = mkfrag(xr2[g * 2 + hi], ZQ);              // K 32..47 (t), pad
        CFrag acc;
        acc.v = wmma_f16(a0, ldfrag(fb + (g * 2 + 0) * 64), v8f{});
        acc.v = wmma_f16(a1, ldfrag(fb + (g * 2 + 1) * 64), acc.v);
        float cb = cbv[g];
        #pragma unroll
        for (int r = 0; r < 8; ++r)
          yb[(r + (hi << 3)) * 64 + g * 16 + nl] = acc.f[r] + cb;
      }
    }

    // ---- LayerNorm (one lane per batch row), write packed f16 -------------
    if (lane < 16) {
      float4 vv[16];
      float s = 0.f, ss = 0.f;
      #pragma unroll
      for (int i = 0; i < 16; ++i) {
        float4 v = ybq[lane][i]; vv[i] = v;
        s  += v.x + v.y + v.z + v.w;
        ss += v.x * v.x + v.y * v.y + v.z * v.z + v.w * v.w;
      }
      float mu = s * (1.f / FEATD);
      float rs = __frsqrt_rn(ss * (1.f / FEATD) - mu * mu + 1e-5f);
      U2* yw = (U2*)&yhq[lane][0];
      #pragma unroll
      for (int i = 0; i < 16; ++i) {
        float4 g4 = sgamq[i], b4 = sbetq[i];
        PK p;
        p.h[0] = (_Float16)((vv[i].x - mu) * rs * g4.x + b4.x);
        p.h[1] = (_Float16)((vv[i].y - mu) * rs * g4.y + b4.y);
        p.h[2] = (_Float16)((vv[i].z - mu) * rs * g4.z + b4.z);
        p.h[3] = (_Float16)((vv[i].w - mu) * rs * g4.w + b4.w);
        yw[i] = p.u;
      }
    }

    // ---- GRU1 -------------------------------------------------------------
    CFrag xg[6], hg[6];
    #pragma unroll
    for (int tn = 0; tn < 6; ++tn) {
      #pragma unroll
      for (int r = 0; r < 8; ++r) { xg[tn].f[r] = rbi1[tn]; hg[tn].f[r] = rbh1[tn]; }
    }
    {
      const U4* yr = &yhq[nl][0];
      #pragma unroll
      for (int ck = 0; ck < 2; ++ck) {
        v16h a = mkfrag(yr[ck * 4 + hi], yr[ck * 4 + 2 + hi]);
        #pragma unroll
        for (int tn = 0; tn < 6; ++tn)
          xg[tn].v = wmma_f16(a, ldfrag(fb + (8 + tn * 2 + ck) * 64), xg[tn].v);
      }
      const U4* hr = &hb1q[nl][0];
      v16h ah = mkfrag(hr[hi], hr[2 + hi]);
      #pragma unroll
      for (int tn = 0; tn < 6; ++tn)
        hg[tn].v = wmma_f16(ah, ldfrag(fb + (20 + tn) * 64), hg[tn].v);
    }
    #pragma unroll
    for (int tile = 0; tile < 2; ++tile)
      #pragma unroll
      for (int r = 0; r < 8; ++r) {
        float rr = sigf(xg[tile].f[r] + hg[tile].f[r]);
        float zz = sigf(xg[2 + tile].f[r] + hg[2 + tile].f[r]);
        float nn = tanhfast(xg[4 + tile].f[r] + rr * hg[4 + tile].f[r]);
        h1c[tile].f[r] = (1.f - zz) * nn + zz * h1c[tile].f[r];
      }
    #pragma unroll
    for (int r = 0; r < 8; ++r) {
      h1p[(r + (hi << 3)) * 32 + nl]      = (_Float16)h1c[0].f[r];
      h1p[(r + (hi << 3)) * 32 + 16 + nl] = (_Float16)h1c[1].f[r]; // cols 24..31: finite pad
    }

    // ---- GRU2 -------------------------------------------------------------
    #pragma unroll
    for (int tn = 0; tn < 6; ++tn) {
      #pragma unroll
      for (int r = 0; r < 8; ++r) { xg[tn].f[r] = rbi2[tn]; hg[tn].f[r] = rbh2[tn]; }
    }
    {
      const U4* h1r = &hb1q[nl][0];
      const U4* h2r = &hb2q[nl][0];
      v16h a1 = mkfrag(h1r[hi], h1r[2 + hi]);   // out1_t
      v16h a2 = mkfrag(h2r[hi], h2r[2 + hi]);   // h2_{t-1}
      #pragma unroll
      for (int tn = 0; tn < 6; ++tn) {
        xg[tn].v = wmma_f16(a1, ldfrag(fb + (26 + tn) * 64), xg[tn].v);
        hg[tn].v = wmma_f16(a2, ldfrag(fb + (32 + tn) * 64), hg[tn].v);
      }
    }
    #pragma unroll
    for (int tile = 0; tile < 2; ++tile)
      #pragma unroll
      for (int r = 0; r < 8; ++r) {
        float rr = sigf(xg[tile].f[r] + hg[tile].f[r]);
        float zz = sigf(xg[2 + tile].f[r] + hg[2 + tile].f[r]);
        float nn = tanhfast(xg[4 + tile].f[r] + rr * hg[4 + tile].f[r]);
        h2c[tile].f[r] = (1.f - zz) * nn + zz * h2c[tile].f[r];
      }
    #pragma unroll
    for (int r = 0; r < 8; ++r) {
      h2p[(r + (hi << 3)) * 32 + nl]      = (_Float16)h2c[0].f[r];
      h2p[(r + (hi << 3)) * 32 + 16 + nl] = (_Float16)h2c[1].f[r];
    }

    // ---- FC head ----------------------------------------------------------
    {
      const U4* h2r = &hb2q[nl][0];
      v16h a = mkfrag(h2r[hi], h2r[2 + hi]);
      CFrag o0, o1;
      #pragma unroll
      for (int r = 0; r < 8; ++r) { o0.f[r] = fcb0; o1.f[r] = fcb1; }
      o0.v = wmma_f16(a, ldfrag(fb + 38 * 64), o0.v);
      o1.v = wmma_f16(a, ldfrag(fb + 39 * 64), o1.v);
      #pragma unroll
      for (int r = 0; r < 8; ++r) {
        int b = r + (hi << 3);
        size_t base = (((size_t)(b0 + b)) * T_LEN + t) * 32;
        out[base + nl]      = o0.f[r];
        out[base + 16 + nl] = o1.f[r];
        nb[b * 32 + nl]      = o0.f[r];
        nb[b * 32 + 16 + nl] = o1.f[r];
      }
    }
    // ---- confidence = 1/(1+std), ddof=1 over 32 outputs -------------------
    if (lane < 16) {
      float s = 0.f, ss = 0.f;
      #pragma unroll
      for (int i = 0; i < 8; ++i) {
        float4 v = nbq[lane][i];
        s  += v.x + v.y + v.z + v.w;
        ss += v.x * v.x + v.y * v.y + v.z * v.z + v.w * v.w;
      }
      float mu = s * (1.f / 32.f);
      float var = (ss - 32.f * mu * mu) * (1.f / 31.f);
      var = var > 0.f ? var : 0.f;
      out[CONF_OFF + (size_t)(b0 + lane) * T_LEN + t] = __frcp_rn(1.f + __fsqrt_rn(var));
    }

    // rotate time slices (scalar only)
    s0 = s1; s1 = s2; s2 = (s2 == 2) ? 0 : (s2 + 1);
  }

  // ---- final hidden states ----
  if (lane < 16)
    for (int u = 0; u < HID; ++u) {
      out[H1_OFF + (size_t)(b0 + lane) * HID + u] = (float)h1p[lane * 32 + u];
      out[H2_OFF + (size_t)(b0 + lane) * HID + u] = (float)h2p[lane * 32 + u];
    }
}

extern "C" void kernel_launch(void* const* d_in, const int* in_sizes, int n_in,
                              void* d_out, int out_size, void* d_ws, size_t ws_size,
                              hipStream_t stream) {
  (void)in_sizes; (void)n_in; (void)out_size; (void)d_ws; (void)ws_size;
  const float* x    = (const float*)d_in[0];
  const float* h1   = (const float*)d_in[1];
  const float* h2   = (const float*)d_in[2];
  const float* cw   = (const float*)d_in[3];
  const float* cb   = (const float*)d_in[4];
  const float* gm   = (const float*)d_in[5];
  const float* bt   = (const float*)d_in[6];
  const float* wih1 = (const float*)d_in[7];
  const float* whh1 = (const float*)d_in[8];
  const float* bih1 = (const float*)d_in[9];
  const float* bhh1 = (const float*)d_in[10];
  const float* wih2 = (const float*)d_in[11];
  const float* whh2 = (const float*)d_in[12];
  const float* bih2 = (const float*)d_in[13];
  const float* bhh2 = (const float*)d_in[14];
  const float* fcw  = (const float*)d_in[15];
  const float* fcbp = (const float*)d_in[16];
  tiny_gru_persist<<<dim3(B_TOT / 16), dim3(32), 0, stream>>>(
      x, h1, h2, cw, cb, gm, bt,
      wih1, whh1, bih1, bhh1,
      wih2, whh2, bih2, bhh2,
      fcw, fcbp, (float*)d_out);
}
